// BertSelfAttention_82368882803320
// MI455X (gfx1250) — compile-verified, matrix operand order checked by reading, not run on hardware
//
#include <hip/hip_runtime.h>
#include <hip/hip_bf16.h>
#include <stdint.h>

#define B_    4
#define N_    4096
#define HID_  768
#define H_    12
#define DH_   64
#define NB_   266
#define NBP_  288            // NB padded to multiple of 32
#define TOK_  (B_*N_)
#define BH_   (B_*H_)

#define RATIO_ 0.06131393691f   // 266^-0.5
#define DN_    0.35355339059f   // 64^-0.25  (folded into proj)
#define DIAGC_ 0.0625f          // 0.5 * dn^2
#define EPS_   1e-4f

typedef __attribute__((ext_vector_type(16))) __bf16 bf16x16;
typedef __attribute__((ext_vector_type(8)))  float  f32x8;

struct U16x32B { uint4 lo; uint4 hi; };

static __device__ __forceinline__ bf16x16 mk_frag(uint4 lo, uint4 hi) {
  U16x32B t; t.lo = lo; t.hi = hi;
  return __builtin_bit_cast(bf16x16, t);
}

static __device__ __forceinline__ f32x8 wmma_bf16(bf16x16 a, bf16x16 b, f32x8 c) {
  // D = A(16x32 bf16) * B(32x16 bf16) + C(16x16 f32)
  return __builtin_amdgcn_wmma_f32_16x16x32_bf16(false, a, false, b, (short)0, c, false, false);
}

static __device__ __forceinline__ f32x8 zero8() {
  f32x8 z;
#pragma unroll
  for (int i = 0; i < 8; ++i) z[i] = 0.f;
  return z;
}

static __device__ __forceinline__ unsigned short f2bf(float f) {
  unsigned int i = __builtin_bit_cast(unsigned int, f);
  unsigned int r = (i + 0x7FFFu + ((i >> 16) & 1u)) >> 16;
  return (unsigned short)r;
}
static __device__ __forceinline__ float bf2f(unsigned short u) {
  unsigned int i = ((unsigned int)u) << 16;
  return __builtin_bit_cast(float, i);
}

// intra-wave LDS RAW fence (DS ops are in-order per wave on CDNA5)
#define DS_FENCE() asm volatile("s_wait_dscnt 0" ::: "memory")

// LDS byte address: flat LDS address truncated to 32 bits == LDS offset (ISA 10.2)
static __device__ __forceinline__ unsigned lds_addr_of(const void* p) {
  return (unsigned)(__SIZE_TYPE__)p;
}

// ---------------- CDNA5 async copy to LDS (ASYNCcnt path) ------------------
#if __has_builtin(__builtin_amdgcn_global_load_async_to_lds_b128)
#define ASYNC_LDS_OK 1
#warning CDNA5_PROBE: global_load_async_to_lds_b128 builtin PRESENT
#else
#define ASYNC_LDS_OK 0
#warning CDNA5_PROBE: global_load_async_to_lds_b128 builtin ABSENT
#endif
#if __has_builtin(__builtin_amdgcn_s_wait_asynccnt)
#warning CDNA5_PROBE: s_wait_asynccnt builtin PRESENT
#else
#warning CDNA5_PROBE: s_wait_asynccnt builtin ABSENT
#endif

typedef __attribute__((ext_vector_type(4))) int v4i_;
typedef __attribute__((address_space(1))) v4i_* g_v4p;
typedef __attribute__((address_space(3))) v4i_* l_v4p;

static __device__ __forceinline__ void async_copy16(unsigned short* lds_dst,
                                                    const unsigned short* gsrc) {
#if ASYNC_LDS_OK
  // go through integers: inttoptr carries no null-check semantics (avoids
  // the cndmask chain an addrspacecast would generate)
  __builtin_amdgcn_global_load_async_to_lds_b128(
      (g_v4p)(__SIZE_TYPE__)gsrc,
      (l_v4p)(__SIZE_TYPE__)lds_addr_of(lds_dst), 0, 0);
#else
  *(uint4*)lds_dst = *(const uint4*)gsrc;
#endif
}
static __device__ __forceinline__ void wait_async0() {
#if ASYNC_LDS_OK
#if __has_builtin(__builtin_amdgcn_s_wait_asynccnt)
  __builtin_amdgcn_s_wait_asynccnt(0);
#else
  asm volatile("s_wait_asynccnt 0x0" ::: "memory");
#endif
#endif
}

// ---------------- Tensor Data Mover (TENSORcnt path) -----------------------
#if __has_builtin(__builtin_amdgcn_tensor_load_to_lds)
#define TDM_OK 1
#warning CDNA5_PROBE: tensor_load_to_lds builtin PRESENT
#else
#define TDM_OK 0
#warning CDNA5_PROBE: tensor_load_to_lds builtin ABSENT
#endif

#if TDM_OK
typedef __attribute__((ext_vector_type(4))) unsigned int u32x4;
typedef __attribute__((ext_vector_type(4))) int i32x4;
typedef __attribute__((ext_vector_type(8))) int i32x8;

// 2D tile of 2-byte elements: rows x cols, row stride in elements.
static __device__ __forceinline__ void tdm_load_2d_bf16(
    const unsigned short* gsrc, unsigned lds_byte_addr,
    unsigned rows, unsigned cols, unsigned row_stride) {
  unsigned long long ga = (unsigned long long)(__SIZE_TYPE__)gsrc;
  u32x4 g0;
  g0[0] = 1u;                                        // count=1, user mode
  g0[1] = lds_byte_addr;                             // D#.lds_addr
  g0[2] = (unsigned)(ga & 0xffffffffu);              // global_addr[31:0]
  g0[3] = (unsigned)((ga >> 32) & 0x01ffffffu) | 0x80000000u;  // addr[56:32], type=2
  i32x8 g1;
  g1[0] = (int)(1u << 16);                           // workgroup_mask=0, data_size=2B
  g1[1] = (int)((cols & 0xffffu) << 16);             // tensor_dim0[15:0]
  g1[2] = (int)(((cols >> 16) & 0xffffu) | ((rows & 0xffffu) << 16)); // dim0 hi | dim1 lo
  g1[3] = (int)(((rows >> 16) & 0xffffu) | ((cols & 0xffffu) << 16)); // dim1 hi | tile_dim0
  g1[4] = (int)(rows & 0xffffu);                     // tile_dim1 (tile_dim2=0)
  g1[5] = (int)row_stride;                           // tensor_dim0_stride[31:0]
  g1[6] = 0;                                         // stride hi | dim1_stride lo
  g1[7] = 0;
  i32x4 z4 = {0, 0, 0, 0};
#if __clang_major__ >= 23
  i32x8 z8 = {0, 0, 0, 0, 0, 0, 0, 0};
  __builtin_amdgcn_tensor_load_to_lds(g0, g1, z4, z4, z8, 0);
#else
  __builtin_amdgcn_tensor_load_to_lds(g0, g1, z4, z4, 0);
#endif
}
#endif

// ---------------------------------------------------------------------------
// K0: fp32 -> bf16 packing (hs, Wq|Wk|Wv, proj*dn with zero padding to NBP)
// ---------------------------------------------------------------------------
__global__ __launch_bounds__(256) void pack_kernel(
    const float* __restrict__ hs, const float* __restrict__ Wq,
    const float* __restrict__ Wk, const float* __restrict__ Wv,
    const float* __restrict__ proj,
    unsigned short* __restrict__ hsb, unsigned short* __restrict__ wb,
    unsigned short* __restrict__ projb) {
  const int n1 = TOK_ * HID_;
  const int n2 = HID_ * HID_;
  const int n3 = NBP_ * DH_;
  int idx = blockIdx.x * 256 + threadIdx.x;
  if (idx < n1) { hsb[idx] = f2bf(hs[idx]); return; }
  idx -= n1;
  if (idx < 3 * n2) {
    int g = idx / n2, r = idx - g * n2;
    const float* s = (g == 0) ? Wq : ((g == 1) ? Wk : Wv);
    wb[idx] = f2bf(s[r]);
    return;
  }
  idx -= 3 * n2;
  if (idx < n3) {
    int row = idx >> 6;
    int d = idx & 63;
    projb[idx] = (row < NB_) ? f2bf(proj[row * DH_ + d] * DN_) : (unsigned short)0;
  }
}

__global__ __launch_bounds__(256) void zero_kernel(float* __restrict__ p, int n) {
  int i = blockIdx.x * 256 + threadIdx.x;
  if (i < n) p[i] = 0.f;
}

// ---------------------------------------------------------------------------
// K1: QKV = hs @ W^T + bias.  Double-buffered LDS W tile staged with
// GLOBAL_LOAD_ASYNC_TO_LDS_B128; one barrier per K-step.
// grid (TOK/128, 2304/64), block 256 (8 waves, 16 tokens x 64 outputs each)
// ---------------------------------------------------------------------------
__global__ __launch_bounds__(256) void qkv_kernel(
    const unsigned short* __restrict__ hsb, const unsigned short* __restrict__ wb,
    const float* __restrict__ bq, const float* __restrict__ bk, const float* __restrict__ bv,
    unsigned short* __restrict__ qb, unsigned short* __restrict__ kb,
    unsigned short* __restrict__ vb) {
  __shared__ alignas(16) unsigned short wtile[2 * 64 * 32];
  const int tid  = threadIdx.x;
  const int wid  = tid >> 5;
  const int l    = tid & 31;
  const int half = l >> 4;
  const int ln   = l & 15;

  const int obase = blockIdx.y * 64;
  const int g     = obase / HID_;
  const int oin0  = obase - g * HID_;
  const int tbase = blockIdx.x * 128;
  const int wtb   = tbase + wid * 16;

  // staging coords: one b128 per thread per tile
  const int s_o  = tid >> 2;
  const int s_ch = (tid & 3) << 3;
  const unsigned short* wrow = wb + (size_t)(g * HID_ + oin0 + s_o) * HID_ + s_ch;
  unsigned short* lds_st = &wtile[s_o * 32 + s_ch];   // + cur*2048

  f32x8 acc[4];
#pragma unroll
  for (int c = 0; c < 4; ++c) acc[c] = zero8();

  const unsigned short* arow = hsb + (size_t)(wtb + ln) * HID_;

  async_copy16(lds_st, wrow);                  // preload first W tile
  int cur = 0;
  for (int kk = 0; kk < HID_; kk += 32) {
    wait_async0();
    __syncthreads();                            // tile[cur] ready for everyone
    if (kk + 32 < HID_) async_copy16(lds_st + (cur ^ 1) * 2048, wrow + kk + 32);
    if (kk + 64 < HID_) __builtin_prefetch(arow + kk + 64, 0, 1);

    uint4 a0 = *(const uint4*)(arow + kk + half * 8);
    uint4 a1 = *(const uint4*)(arow + kk + 16 + half * 8);
    bf16x16 A = mk_frag(a0, a1);
    const unsigned short* wt = &wtile[cur * 2048];
#pragma unroll
    for (int c = 0; c < 4; ++c) {
      int o_l = c * 16 + ln;
      uint4 b0 = *(const uint4*)(wt + o_l * 32 + half * 8);
      uint4 b1 = *(const uint4*)(wt + o_l * 32 + 16 + half * 8);
      acc[c] = wmma_bf16(A, mk_frag(b0, b1), acc[c]);
    }
    cur ^= 1;
  }

  const float* bias  = (g == 0) ? bq : ((g == 1) ? bk : bv);
  unsigned short* dst = (g == 0) ? qb : ((g == 1) ? kb : vb);
#pragma unroll
  for (int c = 0; c < 4; ++c) {
    int oin = oin0 + c * 16 + ln;
    int h   = oin >> 6;
    int d   = oin & 63;
    float bs = bias[oin];
#pragma unroll
    for (int r = 0; r < 8; ++r) {
      int token = wtb + r + half * 8;
      int b = token >> 12;
      int n = token & (N_ - 1);
      dst[((size_t)(b * H_ + h) * N_ + n) * DH_ + d] = f2bf(acc[c][r] + bs);
    }
  }
}

// ---------------------------------------------------------------------------
// K2: per-(b,h) global max of (k*dn)@proj^T (masked j<266) -> block partials
// grid (N/64, BH), block 128 (4 waves x 16 tokens)
// ---------------------------------------------------------------------------
__global__ __launch_bounds__(128) void kmax_kernel(
    const unsigned short* __restrict__ kb, const unsigned short* __restrict__ projb,
    float* __restrict__ partial) {
  __shared__ float red[4];
  const int tid = threadIdx.x;
  const int wid = tid >> 5;
  const int l = tid & 31, half = l >> 4, ln = l & 15;
  const int bh = blockIdx.y;
  const int tb = blockIdx.x * 64 + wid * 16;

  const unsigned short* arow = kb + ((size_t)bh * N_ + tb + ln) * DH_;
  bf16x16 A0 = mk_frag(*(const uint4*)(arow + half * 8),
                       *(const uint4*)(arow + 16 + half * 8));
  bf16x16 A1 = mk_frag(*(const uint4*)(arow + 32 + half * 8),
                       *(const uint4*)(arow + 48 + half * 8));

  float m = -1e30f;
  for (int jt = 0; jt < 17; ++jt) {   // jt==17 is fully padding: skip
    const unsigned short* brow = projb + (size_t)(jt * 16 + ln) * DH_;
    bf16x16 B0 = mk_frag(*(const uint4*)(brow + half * 8),
                         *(const uint4*)(brow + 16 + half * 8));
    bf16x16 B1 = mk_frag(*(const uint4*)(brow + 32 + half * 8),
                         *(const uint4*)(brow + 48 + half * 8));
    f32x8 c = zero8();
    c = wmma_bf16(A0, B0, c);
    c = wmma_bf16(A1, B1, c);
    bool valid = (jt < 16) || (ln < 10);   // j = jt*16 + ln < 266
    if (valid) {
#pragma unroll
      for (int r = 0; r < 8; ++r) m = fmaxf(m, c[r]);
    }
  }
#pragma unroll
  for (int off = 16; off > 0; off >>= 1) m = fmaxf(m, __shfl_xor(m, off, 32));
  if (l == 0) red[wid] = m;
  __syncthreads();
  if (tid == 0) {
    partial[bh * 64 + blockIdx.x] =
        fmaxf(fmaxf(red[0], red[1]), fmaxf(red[2], red[3]));
  }
}

__global__ __launch_bounds__(64) void kmax_reduce_kernel(
    const float* __restrict__ partial, float* __restrict__ kmax) {
  __shared__ float red[64];
  int bh = blockIdx.x;
  red[threadIdx.x] = partial[bh * 64 + threadIdx.x];
  __syncthreads();
  for (int s = 32; s > 0; s >>= 1) {
    if (threadIdx.x < s) red[threadIdx.x] = fmaxf(red[threadIdx.x], red[threadIdx.x + s]);
    __syncthreads();
  }
  if (threadIdx.x == 0) kmax[bh] = red[0];
}

// ---------------------------------------------------------------------------
// K4: ctx[j,d] += sum_n kp[n,j]*v[n,d]; ksum[j] += sum_n kp[n,j]
// k tile staged by the Tensor Data Mover (TENSORcnt), v tile transposed by hand.
// grid (4, BH), block 576 (18 waves, one 16-row NB tile each); 32-token steps
// ---------------------------------------------------------------------------
__global__ __launch_bounds__(576) void ctx_kernel(
    const unsigned short* __restrict__ kb, const unsigned short* __restrict__ vb,
    const unsigned short* __restrict__ projb, const float* __restrict__ kmax,
    float* __restrict__ ctx, float* __restrict__ ksum) {
  __shared__ alignas(16) unsigned short kbuf[32 * 64];
  __shared__ alignas(16) unsigned short vTb[64 * 32];
  __shared__ alignas(16) unsigned short kpbuf[18][16 * 32];
  __shared__ float diagk[32];

  const int tid = threadIdx.x;
  const int wid = tid >> 5;            // 0..17 -> NB tile
  const int l = tid & 31, half = l >> 4, ln = l & 15;
  const int bh  = blockIdx.y;
  const int nb0 = blockIdx.x * 1024;
  const float m = kmax[bh];

  // fixed A = proj rows (j tile), K = 64 -> two chunks
  const unsigned short* prow = projb + (size_t)(wid * 16 + ln) * DH_;
  bf16x16 Ap0 = mk_frag(*(const uint4*)(prow + half * 8),
                        *(const uint4*)(prow + 16 + half * 8));
  bf16x16 Ap1 = mk_frag(*(const uint4*)(prow + 32 + half * 8),
                        *(const uint4*)(prow + 48 + half * 8));

  f32x8 cacc[4];
#pragma unroll
  for (int c = 0; c < 4; ++c) cacc[c] = zero8();
  float ksacc[8];
#pragma unroll
  for (int r = 0; r < 8; ++r) ksacc[r] = 0.f;

  const unsigned kbuf_lds = lds_addr_of(&kbuf[0]);

  for (int ks = 0; ks < 32; ++ks) {
    const int nbase = nb0 + ks * 32;
    __syncthreads();                       // previous tile fully consumed
#if TDM_OK
    if (wid == 0) {
      // TDM: 32x64 bf16 tile, row stride 64 elements, straight into LDS
      tdm_load_2d_bf16(kb + ((size_t)bh * N_ + nbase) * DH_, kbuf_lds, 32, 64, 64);
    }
#else
    if (tid < 256) {
      int n_l = tid >> 3, ch = (tid & 7) << 3;
      *(uint4*)&kbuf[n_l * 64 + ch] =
          *(const uint4*)(kb + ((size_t)bh * N_ + nbase + n_l) * DH_ + ch);
    }
#endif
    if (tid >= 256 && tid < 512) {         // stage v tile transposed [64 x 32]
      int t2 = tid - 256;
      int n_l = t2 >> 3, ch = (t2 & 7) << 3;
      union { uint4 v; unsigned short s[8]; } u;
      u.v = *(const uint4*)(vb + ((size_t)bh * N_ + nbase + n_l) * DH_ + ch);
#pragma unroll
      for (int e = 0; e < 8; ++e) vTb[(ch + e) * 32 + n_l] = u.s[e];
    }
#if TDM_OK
    if (wid == 0) __builtin_amdgcn_s_wait_tensorcnt(0);
#endif
    __syncthreads();
    if (tid < 32) {                        // diag per token
      float s = 0.f;
      for (int d = 0; d < 64; ++d) { float x = bf2f(kbuf[tid * 64 + d]); s += x * x; }
      diagk[tid] = s * DIAGC_;
    }
    __syncthreads();

    // dash^T = proj @ k^T for two 16-token column tiles, then exp -> kp
#pragma unroll
    for (int nt = 0; nt < 2; ++nt) {
      const unsigned short* krow = &kbuf[(nt * 16 + ln) * 64];
      bf16x16 Bk0 = mk_frag(*(const uint4*)(krow + half * 8),
                            *(const uint4*)(krow + 16 + half * 8));
      bf16x16 Bk1 = mk_frag(*(const uint4*)(krow + 32 + half * 8),
                            *(const uint4*)(krow + 48 + half * 8));
      f32x8 c = zero8();
      c = wmma_bf16(Ap0, Bk0, c);
      c = wmma_bf16(Ap1, Bk1, c);
      int tl = nt * 16 + ln;
      float dg = diagk[tl];
#pragma unroll
      for (int r = 0; r < 8; ++r) {
        int j = wid * 16 + r + half * 8;
        float kp = 0.f;
        if (j < NB_) kp = RATIO_ * (__expf(c[r] - dg - m) + EPS_);
        ksacc[r] += kp;
        kpbuf[wid][(r + half * 8) * 32 + tl] = f2bf(kp);   // C-layout -> A-layout
      }
    }
    DS_FENCE();

    // ctx += kp(16j x 32n) @ v(32n x 64d)
    const unsigned short* krow2 = &kpbuf[wid][ln * 32];
    bf16x16 Akp = mk_frag(*(const uint4*)(krow2 + half * 8),
                          *(const uint4*)(krow2 + 16 + half * 8));
#pragma unroll
    for (int dt = 0; dt < 4; ++dt) {
      const unsigned short* vrow = &vTb[(dt * 16 + ln) * 32];
      bf16x16 Bv = mk_frag(*(const uint4*)(vrow + half * 8),
                           *(const uint4*)(vrow + 16 + half * 8));
      cacc[dt] = wmma_bf16(Akp, Bv, cacc[dt]);
    }
  }

  // reduce + accumulate globally
#pragma unroll
  for (int r = 0; r < 8; ++r) {
    float s = ksacc[r];
#pragma unroll
    for (int off = 8; off > 0; off >>= 1) s += __shfl_xor(s, off, 16);
    if (ln == 0) {
      int j = wid * 16 + r + half * 8;
      unsafeAtomicAdd(&ksum[bh * NBP_ + j], s);
    }
  }
#pragma unroll
  for (int dt = 0; dt < 4; ++dt) {
#pragma unroll
    for (int r = 0; r < 8; ++r) {
      int j = wid * 16 + r + half * 8;
      int d = dt * 16 + ln;
      unsafeAtomicAdd(&ctx[((size_t)bh * NBP_ + j) * DH_ + d], cacc[dt][r]);
    }
  }
}

// ---------------------------------------------------------------------------
// K5: qp = featuremap(q); out = (qp @ ctx) / (qp . ksum)
// grid (N/32, BH), block 64 (2 waves x 16 tokens)
// ---------------------------------------------------------------------------
__global__ __launch_bounds__(64) void out_kernel(
    const unsigned short* __restrict__ qb, const unsigned short* __restrict__ projb,
    const float* __restrict__ ctx, const float* __restrict__ ksum,
    float* __restrict__ out) {
  __shared__ alignas(16) unsigned short ctxT[64 * NBP_];
  __shared__ alignas(16) unsigned short qpbuf[2][16 * NBP_];
  __shared__ float ksums[NBP_];
  __shared__ float diagq[2][16];
  __shared__ float qmaxl[2][16];
  __shared__ float dinvl[2][16];

  const int tid = threadIdx.x;
  const int wid = tid >> 5;
  const int l = tid & 31, half = l >> 4, ln = l & 15;
  const int bh = blockIdx.y;
  const int b = bh / H_;
  const int h = bh - b * H_;
  const int tb = blockIdx.x * 32 + wid * 16;

  for (int idx = tid; idx < NBP_ * DH_; idx += 64) {   // stage ctx^T as bf16
    int j = idx >> 6, d = idx & 63;
    ctxT[d * NBP_ + j] = f2bf(ctx[(size_t)bh * NBP_ * DH_ + idx]);
  }
  for (int idx = tid; idx < NBP_; idx += 64) ksums[idx] = ksum[bh * NBP_ + idx];
  __syncthreads();

  const unsigned short* arow = qb + ((size_t)bh * N_ + tb + ln) * DH_;
  bf16x16 Aq0 = mk_frag(*(const uint4*)(arow + half * 8),
                        *(const uint4*)(arow + 16 + half * 8));
  bf16x16 Aq1 = mk_frag(*(const uint4*)(arow + 32 + half * 8),
                        *(const uint4*)(arow + 48 + half * 8));

  if (l < 16) {
    const unsigned short* qrow = qb + ((size_t)bh * N_ + tb + l) * DH_;
    float s = 0.f;
    for (int d = 0; d < 64; ++d) { float x = bf2f(qrow[d]); s += x * x; }
    diagq[wid][l] = s * DIAGC_;
  }

  float maxacc[8];
#pragma unroll
  for (int r = 0; r < 8; ++r) maxacc[r] = -1e30f;

  for (int jt = 0; jt < 18; ++jt) {       // dash tiles -> LDS, masked row max
    const unsigned short* brow = projb + (size_t)(jt * 16 + ln) * DH_;
    bf16x16 B0 = mk_frag(*(const uint4*)(brow + half * 8),
                         *(const uint4*)(brow + 16 + half * 8));
    bf16x16 B1 = mk_frag(*(const uint4*)(brow + 32 + half * 8),
                         *(const uint4*)(brow + 48 + half * 8));
    f32x8 c = zero8();
    c = wmma_bf16(Aq0, B0, c);
    c = wmma_bf16(Aq1, B1, c);
    bool valid = (jt < 16) || ((jt == 16) && (ln < 10));
#pragma unroll
    for (int r = 0; r < 8; ++r) {
      qpbuf[wid][(r + half * 8) * NBP_ + jt * 16 + ln] = f2bf(c[r]);
      if (valid) maxacc[r] = fmaxf(maxacc[r], c[r]);
    }
  }
#pragma unroll
  for (int r = 0; r < 8; ++r) {
    float mr = maxacc[r];
#pragma unroll
    for (int off = 8; off > 0; off >>= 1) mr = fmaxf(mr, __shfl_xor(mr, off, 16));
    if (ln == 0) qmaxl[wid][r + half * 8] = mr;
  }
  DS_FENCE();

  {  // exp + denominator (2 lanes per token row)
    int tl = l >> 1;
    int j0 = (l & 1) * (NBP_ / 2);
    float dg = diagq[wid][tl];
    float mr = qmaxl[wid][tl];
    float dsum = 0.f;
    for (int jj = 0; jj < NBP_ / 2; ++jj) {
      int j = j0 + jj;
      float v = bf2f(qpbuf[wid][tl * NBP_ + j]);
      float qp = RATIO_ * (__expf(v - dg - mr) + EPS_);
      qpbuf[wid][tl * NBP_ + j] = f2bf(qp);
      dsum += qp * ksums[j];
    }
    dsum += __shfl_xor(dsum, 1, 32);
    if ((l & 1) == 0) dinvl[wid][tl] = 1.0f / dsum;
  }
  DS_FENCE();

  // out tile = qp(16 x 288) @ ctxT(288 x 64), scaled by dinv
#pragma unroll
  for (int dt = 0; dt < 4; ++dt) {
    f32x8 c = zero8();
    const unsigned short* crow = &ctxT[(dt * 16 + ln) * NBP_];
#pragma unroll
    for (int kc = 0; kc < NBP_ / 32; ++kc) {
      const unsigned short* qrow = &qpbuf[wid][ln * NBP_ + kc * 32];
      bf16x16 A = mk_frag(*(const uint4*)(qrow + half * 8),
                          *(const uint4*)(qrow + 16 + half * 8));
      bf16x16 Bc = mk_frag(*(const uint4*)(crow + kc * 32 + half * 8),
                           *(const uint4*)(crow + kc * 32 + 16 + half * 8));
      c = wmma_bf16(A, Bc, c);
    }
#pragma unroll
    for (int r = 0; r < 8; ++r) {
      int tloc = r + half * 8;
      int n = tb + tloc;
      int d = dt * 16 + ln;
      out[((size_t)b * N_ + n) * HID_ + h * DH_ + d] = c[r] * dinvl[wid][tloc];
    }
  }
}

// ---------------------------------------------------------------------------
extern "C" void kernel_launch(void* const* d_in, const int* in_sizes, int n_in,
                              void* d_out, int out_size, void* d_ws, size_t ws_size,
                              hipStream_t stream) {
  const float* hs   = (const float*)d_in[0];
  const float* Wq   = (const float*)d_in[1];
  const float* bq   = (const float*)d_in[2];
  const float* Wk   = (const float*)d_in[3];
  const float* bk   = (const float*)d_in[4];
  const float* Wv   = (const float*)d_in[5];
  const float* bv   = (const float*)d_in[6];
  const float* proj = (const float*)d_in[7];
  float* out = (float*)d_out;

  char* ws = (char*)d_ws;
  size_t o = 0;
  unsigned short* hsb   = (unsigned short*)(ws + o); o += (size_t)TOK_ * HID_ * 2;
  unsigned short* wb    = (unsigned short*)(ws + o); o += (size_t)3 * HID_ * HID_ * 2;
  unsigned short* projb = (unsigned short*)(ws + o); o += (size_t)NBP_ * DH_ * 2;
  unsigned short* qb    = (unsigned short*)(ws + o); o += (size_t)TOK_ * HID_ * 2;
  unsigned short* kb    = (unsigned short*)(ws + o); o += (size_t)TOK_ * HID_ * 2;
  unsigned short* vb    = (unsigned short*)(ws + o); o += (size_t)TOK_ * HID_ * 2;
  float* ctx     = (float*)(ws + o); o += (size_t)BH_ * NBP_ * DH_ * 4;
  float* ksum    = (float*)(ws + o); o += (size_t)BH_ * NBP_ * 4;     // contiguous after ctx
  float* kmax    = (float*)(ws + o); o += (size_t)BH_ * 4;
  float* partial = (float*)(ws + o); o += (size_t)BH_ * 64 * 4;
  (void)ws_size; (void)in_sizes; (void)n_in; (void)out_size;

  int packN = TOK_ * HID_ + 3 * HID_ * HID_ + NBP_ * DH_;
  pack_kernel<<<(packN + 255) / 256, 256, 0, stream>>>(hs, Wq, Wk, Wv, proj,
                                                       hsb, wb, projb);
  int zn = BH_ * NBP_ * DH_ + BH_ * NBP_;   // ctx + ksum (adjacent)
  zero_kernel<<<(zn + 255) / 256, 256, 0, stream>>>(ctx, zn);

  qkv_kernel<<<dim3(TOK_ / 128, (3 * HID_) / 64), 256, 0, stream>>>(
      hsb, wb, bq, bk, bv, qb, kb, vb);
  kmax_kernel<<<dim3(N_ / 64, BH_), 128, 0, stream>>>(kb, projb, partial);
  kmax_reduce_kernel<<<BH_, 64, 0, stream>>>(partial, kmax);
  ctx_kernel<<<dim3(4, BH_), 576, 0, stream>>>(kb, vb, projb, kmax, ctx, ksum);
  out_kernel<<<dim3(N_ / 32, BH_), 64, 0, stream>>>(qb, projb, ctx, ksum, out);
}
